// TinyAttention_74414603371080
// MI455X (gfx1250) — matrix operand order repaired and checked by reading, not compiled, not run on hardware
//
#include <hip/hip_runtime.h>

typedef _Float16 half_t;
typedef __attribute__((ext_vector_type(16))) _Float16 v16h;
typedef __attribute__((ext_vector_type(8)))  _Float16 v8h;
typedef __attribute__((ext_vector_type(8)))  float    v8f;

#define N_SEQ   8192
#define D_MODEL 1024
#define KEY_DIM 64
#define OUT_DIM 1024
#define NEG_INF -1.0e10f

#define WMMA_F16(A, B, C) \
  __builtin_amdgcn_wmma_f32_16x16x32_f16(false, (A), false, (B), (short)0, (C), false, false)

union V16U { v16h v; v8h h[2]; };

// A-fragment (16x32 f16): lane m = lane%16, group g = lane/16.
// halves 0..7  = row[k0 + g*8 .. +7], halves 8..15 = row[k0 + 16 + g*8 .. +7]
__device__ inline v16h load_afrag_f16(const half_t* p, int row, int ld, int k0, int g) {
  V16U r;
  const half_t* base = p + (size_t)row * ld + k0 + g * 8;
  r.h[0] = *(const v8h*)(base);
  r.h[1] = *(const v8h*)(base + 16);
  return r.v;
}

// A-fragment pair (hi/lo split) sourced from an f32 matrix
__device__ inline void load_afrag_f32_hilo(const float* p, int row, int ld, int k0, int g,
                                           v16h* hi, v16h* lo) {
  const float* q0 = p + (size_t)row * ld + k0 + g * 8;
  const float* q1 = q0 + 16;
  V16U h, l;
#pragma unroll
  for (int i = 0; i < 8; ++i) {
    float a = q0[i], b = q1[i];
    half_t ah = (half_t)a, bh = (half_t)b;
    h.v[i] = ah;                       h.v[8 + i] = bh;
    l.v[i] = (half_t)(a - (float)ah);  l.v[8 + i] = (half_t)(b - (float)bh);
  }
  *hi = h.v; *lo = l.v;
}

// B-fragment (32x16 f16) from Bt (row-major transpose of B): lane n = lane%16,
// group g = lane/16. halves 0..7 = Bt[n][k0 + g*8 .. +7], halves 8..15 = Bt[n][k0 + 16 + g*8 .. +7]
__device__ inline v16h load_bfrag_f16(const half_t* p, int row, int ld, int k0, int g) {
  V16U r;
  const half_t* base = p + (size_t)row * ld + k0 + g * 8;
  r.h[0] = *(const v8h*)(base);
  r.h[1] = *(const v8h*)(base + 16);
  return r.v;
}

__device__ inline v16h ones_frag() {
  V16U r;
#pragma unroll
  for (int i = 0; i < 16; ++i) r.v[i] = (half_t)1.0f;
  return r.v;
}

// ---- prep: transpose weights to f16 hi/lo ----
__global__ void prep_wqkv_kernel(const float* __restrict__ w,
                                 half_t* __restrict__ wT_hi, half_t* __restrict__ wT_lo) {
  int i = blockIdx.x * blockDim.x + threadIdx.x;      // i over [3*KEY_DIM][D_MODEL]
  if (i >= 3 * KEY_DIM * D_MODEL) return;
  int n = i / D_MODEL, k = i % D_MODEL;
  float v = w[(size_t)k * (3 * KEY_DIM) + n];
  half_t h = (half_t)v;
  wT_hi[i] = h;
  wT_lo[i] = (half_t)(v - (float)h);
}

__global__ void prep_wout_kernel(const float* __restrict__ w,
                                 half_t* __restrict__ wT_hi, half_t* __restrict__ wT_lo) {
  int i = blockIdx.x * blockDim.x + threadIdx.x;      // i over [OUT_DIM][KEY_DIM]
  if (i >= OUT_DIM * KEY_DIM) return;
  int n = i / KEY_DIM, k = i % KEY_DIM;
  float v = w[(size_t)k * OUT_DIM + n];
  half_t h = (half_t)v;
  wT_hi[i] = h;
  wT_lo[i] = (half_t)(v - (float)h);
}

// ---- QKV projection: one wave does a 16-row tile x all 12 col-chunks (A reuse) ----
__global__ void __launch_bounds__(256)
qkv_proj_kernel(const float* __restrict__ x,
                const half_t* __restrict__ wT_hi, const half_t* __restrict__ wT_lo,
                const float* __restrict__ bqkv,
                half_t* __restrict__ qh, half_t* __restrict__ ql,
                half_t* __restrict__ kh, half_t* __restrict__ kl,
                half_t* __restrict__ vt) {
  int wave = blockIdx.x * (blockDim.x >> 5) + (threadIdx.x >> 5);
  int lane = threadIdx.x & 31;
  int m0 = wave * 16;
  if (m0 >= N_SEQ) return;
  int ml = lane & 15, g = lane >> 4;

  v8f c[12];
#pragma unroll
  for (int nc = 0; nc < 12; ++nc) c[nc] = (v8f){};

  for (int k0 = 0; k0 < D_MODEL; k0 += 32) {
    v16h ahi, alo;
    load_afrag_f32_hilo(x, m0 + ml, D_MODEL, k0, g, &ahi, &alo);
#pragma unroll
    for (int nc = 0; nc < 12; ++nc) {
      v16h bh = load_bfrag_f16(wT_hi, nc * 16 + ml, D_MODEL, k0, g);
      v16h bl = load_bfrag_f16(wT_lo, nc * 16 + ml, D_MODEL, k0, g);
      c[nc] = WMMA_F16(ahi, bh, c[nc]);
      c[nc] = WMMA_F16(ahi, bl, c[nc]);
      c[nc] = WMMA_F16(alo, bh, c[nc]);
    }
  }
#pragma unroll
  for (int nc = 0; nc < 12; ++nc) {
    float bias = bqkv[nc * 16 + ml];
#pragma unroll
    for (int r = 0; r < 8; ++r) {
      int m = m0 + r + 8 * g;
      float val = c[nc][r] + bias;
      half_t vhi = (half_t)val;
      half_t vlo = (half_t)(val - (float)vhi);
      if (nc < 4) {
        qh[(size_t)m * KEY_DIM + nc * 16 + ml] = vhi;
        ql[(size_t)m * KEY_DIM + nc * 16 + ml] = vlo;
      } else if (nc < 8) {
        kh[(size_t)m * KEY_DIM + (nc - 4) * 16 + ml] = vhi;
        kl[(size_t)m * KEY_DIM + (nc - 4) * 16 + ml] = vlo;
      } else {  // transposed store for the P.V B-operand: vt[d][m]
        vt[(size_t)((nc - 8) * 16 + ml) * N_SEQ + m] = vhi;
      }
    }
  }
}

// ---- fused flash attention: 16 query rows/wave, 32 keys/iter, hi-lo logits ----
__global__ void __launch_bounds__(128)
flash_attn_kernel(const half_t* __restrict__ qh, const half_t* __restrict__ ql,
                  const half_t* __restrict__ kh, const half_t* __restrict__ kl,
                  const half_t* __restrict__ vt,
                  const int* __restrict__ causal_p,
                  half_t* __restrict__ attn_hi, half_t* __restrict__ attn_lo) {
  __shared__ half_t Pst[4][16 * 32];     // per-wave P staging (C-layout -> A-layout)
  int wslot = threadIdx.x >> 5;
  int wave  = blockIdx.x * (blockDim.x >> 5) + wslot;
  int lane  = threadIdx.x & 31;
  int m0 = wave * 16;
  if (m0 >= N_SEQ) return;
  int ml = lane & 15, g = lane >> 4;
  int causal = *causal_p;

  v16h qh0 = load_afrag_f16(qh, m0 + ml, KEY_DIM, 0, g);
  v16h qh1 = load_afrag_f16(qh, m0 + ml, KEY_DIM, 32, g);
  v16h ql0 = load_afrag_f16(ql, m0 + ml, KEY_DIM, 0, g);
  v16h ql1 = load_afrag_f16(ql, m0 + ml, KEY_DIM, 32, g);
  const v16h ones = ones_frag();

  float rowm[8];
#pragma unroll
  for (int r = 0; r < 8; ++r) rowm[r] = -3.0e38f;
  v8f o0 = {}, o1 = {}, o2 = {}, o3 = {}, lsum = {};

  int jend = causal ? (m0 + 16) : N_SEQ;
  for (int jb = 0; jb < jend; jb += 32) {
    // ---- S = Q.K^T (hi/lo 3-term) for keys [jb, jb+31] ----
    v8f s0 = {}, s1 = {};
#pragma unroll
    for (int sub = 0; sub < 2; ++sub) {
      int krow = jb + sub * 16 + ml;
      v16h bh0 = load_bfrag_f16(kh, krow, KEY_DIM, 0, g);
      v16h bl0 = load_bfrag_f16(kl, krow, KEY_DIM, 0, g);
      v16h bh1 = load_bfrag_f16(kh, krow, KEY_DIM, 32, g);
      v16h bl1 = load_bfrag_f16(kl, krow, KEY_DIM, 32, g);
      v8f acc = sub ? s1 : s0;
      acc = WMMA_F16(qh0, bh0, acc);
      acc = WMMA_F16(qh0, bl0, acc);
      acc = WMMA_F16(ql0, bh0, acc);
      acc = WMMA_F16(qh1, bh1, acc);
      acc = WMMA_F16(qh1, bl1, acc);
      acc = WMMA_F16(ql1, bh1, acc);
      if (sub) s1 = acc; else s0 = acc;
    }
    // ---- causal mask (only diagonal-touching blocks) ----
    if (causal && (jb + 31 > m0)) {
#pragma unroll
      for (int r = 0; r < 8; ++r) {
        int m = m0 + r + 8 * g;
        if (jb + ml > m)      s0[r] = NEG_INF;
        if (jb + 16 + ml > m) s1[r] = NEG_INF;
      }
    }
    // ---- online max + rescale (row sums come from the ones-WMMA below) ----
#pragma unroll
    for (int r = 0; r < 8; ++r) {
      float mx = fmaxf(s0[r], s1[r]);
#pragma unroll
      for (int sw = 1; sw < 16; sw <<= 1) mx = fmaxf(mx, __shfl_xor(mx, sw, 32));
      float mnew  = fmaxf(rowm[r], mx);
      float alpha = __expf(rowm[r] - mnew);
      rowm[r] = mnew;
      s0[r] = __expf(s0[r] - mnew);
      s1[r] = __expf(s1[r] - mnew);
      o0[r] *= alpha; o1[r] *= alpha; o2[r] *= alpha; o3[r] *= alpha;
      lsum[r] *= alpha;
    }
    // ---- repack P (C-layout f32) -> A-layout f16 via LDS ----
    half_t* P = Pst[wslot];
#pragma unroll
    for (int r = 0; r < 8; ++r) {
      P[(r + 8 * g) * 32 + ml]      = (half_t)s0[r];
      P[(r + 8 * g) * 32 + 16 + ml] = (half_t)s1[r];
    }
    v16h pa = load_afrag_f16(P, ml, 32, 0, g);   // same-wave LDS: DS in-order
    // ---- row sums: lsum += P @ ones (replaces 32 shuffles) ----
    lsum = WMMA_F16(pa, ones, lsum);
    // ---- O += P . V  (B^T = Vt rows, k-base = jb) ----
    o0 = WMMA_F16(pa, load_bfrag_f16(vt, 0  + ml, N_SEQ, jb, g), o0);
    o1 = WMMA_F16(pa, load_bfrag_f16(vt, 16 + ml, N_SEQ, jb, g), o1);
    o2 = WMMA_F16(pa, load_bfrag_f16(vt, 32 + ml, N_SEQ, jb, g), o2);
    o3 = WMMA_F16(pa, load_bfrag_f16(vt, 48 + ml, N_SEQ, jb, g), o3);
  }
  // ---- normalize, store attn output as f16 hi/lo for the final projection ----
#pragma unroll
  for (int r = 0; r < 8; ++r) {
    float inv = 1.0f / lsum[r];
    int m = m0 + r + 8 * g;
    float vals[4] = { o0[r] * inv, o1[r] * inv, o2[r] * inv, o3[r] * inv };
#pragma unroll
    for (int d = 0; d < 4; ++d) {
      half_t vhi = (half_t)vals[d];
      attn_hi[(size_t)m * KEY_DIM + d * 16 + ml] = vhi;
      attn_lo[(size_t)m * KEY_DIM + d * 16 + ml] = (half_t)(vals[d] - (float)vhi);
    }
  }
}

// ---- final projection: one wave does a 16-row tile x 8 col-chunks (A reuse) ----
__global__ void __launch_bounds__(256)
out_proj_kernel(const half_t* __restrict__ attn_hi, const half_t* __restrict__ attn_lo,
                const half_t* __restrict__ wT_hi, const half_t* __restrict__ wT_lo,
                const float* __restrict__ bout, float* __restrict__ out) {
  int wave = blockIdx.x * (blockDim.x >> 5) + (threadIdx.x >> 5);
  int lane = threadIdx.x & 31;
  int grp = wave & 7;                   // 8 groups of 8 chunks over 1024 cols
  int m0 = (wave >> 3) * 16;
  if (m0 >= N_SEQ) return;
  int ml = lane & 15, g = lane >> 4;

  v16h ah0 = load_afrag_f16(attn_hi, m0 + ml, KEY_DIM, 0, g);
  v16h ah1 = load_afrag_f16(attn_hi, m0 + ml, KEY_DIM, 32, g);
  v16h al0 = load_afrag_f16(attn_lo, m0 + ml, KEY_DIM, 0, g);
  v16h al1 = load_afrag_f16(attn_lo, m0 + ml, KEY_DIM, 32, g);

#pragma unroll
  for (int t = 0; t < 8; ++t) {
    int nchunk = grp * 8 + t;
    v16h bh0 = load_bfrag_f16(wT_hi, nchunk * 16 + ml, KEY_DIM, 0, g);
    v16h bl0 = load_bfrag_f16(wT_lo, nchunk * 16 + ml, KEY_DIM, 0, g);
    v16h bh1 = load_bfrag_f16(wT_hi, nchunk * 16 + ml, KEY_DIM, 32, g);
    v16h bl1 = load_bfrag_f16(wT_lo, nchunk * 16 + ml, KEY_DIM, 32, g);
    v8f c = {};
    c = WMMA_F16(ah0, bh0, c);
    c = WMMA_F16(ah0, bl0, c);
    c = WMMA_F16(al0, bh0, c);
    c = WMMA_F16(ah1, bh1, c);
    c = WMMA_F16(ah1, bl1, c);
    c = WMMA_F16(al1, bh1, c);
    float bias = bout[nchunk * 16 + ml];
#pragma unroll
    for (int r = 0; r < 8; ++r)
      out[(size_t)(m0 + r + 8 * g) * OUT_DIM + nchunk * 16 + ml] = c[r] + bias;
  }
}

extern "C" void kernel_launch(void* const* d_in, const int* in_sizes, int n_in,
                              void* d_out, int out_size, void* d_ws, size_t ws_size,
                              hipStream_t stream) {
  const float* x      = (const float*)d_in[0];
  const float* wqkv   = (const float*)d_in[1];
  const float* bqkv   = (const float*)d_in[2];
  const float* wout   = (const float*)d_in[3];
  const float* bout   = (const float*)d_in[4];
  const int*   causal = (const int*)d_in[5];
  float* out = (float*)d_out;
  (void)in_sizes; (void)n_in; (void)out_size; (void)ws_size;

  // workspace layout (all 256B aligned)
  uint8_t* ws = (uint8_t*)d_ws;
  const size_t SZ_WQKV = (size_t)3 * KEY_DIM * D_MODEL * 2;  // 393216
  const size_t SZ_WOUT = (size_t)OUT_DIM * KEY_DIM * 2;      // 131072
  const size_t SZ_MAT  = (size_t)N_SEQ * KEY_DIM * 2;        // 1 MiB
  size_t off = 0;
  half_t* wqkvT_hi = (half_t*)(ws + off); off += SZ_WQKV;
  half_t* wqkvT_lo = (half_t*)(ws + off); off += SZ_WQKV;
  half_t* woutT_hi = (half_t*)(ws + off); off += SZ_WOUT;
  half_t* woutT_lo = (half_t*)(ws + off); off += SZ_WOUT;
  half_t* qhi  = (half_t*)(ws + off); off += SZ_MAT;
  half_t* qlo  = (half_t*)(ws + off); off += SZ_MAT;
  half_t* khi  = (half_t*)(ws + off); off += SZ_MAT;
  half_t* klo  = (half_t*)(ws + off); off += SZ_MAT;
  half_t* vt   = (half_t*)(ws + off); off += SZ_MAT;          // transposed V (64 x 8192)
  half_t* athi = (half_t*)(ws + off); off += SZ_MAT;
  half_t* atlo = (half_t*)(ws + off); off += SZ_MAT;

  // 1) weight transposes to f16 hi/lo
  prep_wqkv_kernel<<<(3 * KEY_DIM * D_MODEL + 255) / 256, 256, 0, stream>>>(wqkv, wqkvT_hi, wqkvT_lo);
  prep_wout_kernel<<<(OUT_DIM * KEY_DIM + 255) / 256, 256, 0, stream>>>(wout, woutT_hi, woutT_lo);

  // 2) QKV projection: 512 waves (16-row tile each, 12 chunks), 8 waves/block
  qkv_proj_kernel<<<64, 256, 0, stream>>>(x, wqkvT_hi, wqkvT_lo, bqkv, qhi, qlo, khi, klo, vt);

  // 3) fused flash attention: 512 waves, 4 waves/block
  flash_attn_kernel<<<128, 128, 0, stream>>>(qhi, qlo, khi, klo, vt, causal, athi, atlo);

  // 4) output projection: 512 row-tiles * 8 chunk-groups = 4096 waves, 8 waves/block
  out_proj_kernel<<<512, 256, 0, stream>>>(athi, atlo, woutT_hi, woutT_lo, bout, out);
}
